// TransformerBlockQuantum_65481071397685
// MI455X (gfx1250) — matrix operand order repaired and checked by reading, not compiled, not run on hardware
//
#include <hip/hip_runtime.h>

// ---------------------------------------------------------------------------
// CDNA5 (gfx1250) fused transformer block with quantum FFN.
// Heavy GEMMs + attention use v_wmma_f32_16x16x32_bf16 (wave32 WMMA).
// ---------------------------------------------------------------------------

typedef __attribute__((ext_vector_type(16))) __bf16    v16bf;
typedef __attribute__((ext_vector_type(8)))  float     v8f;
typedef __attribute__((ext_vector_type(4)))  unsigned  v4u;

// A-fragment (16x32 bf16, MxK): lane L holds row M = L&15.
//   elems 0..7  = K = kbase + 0..7      (kbase = kk + (L>=16 ? 8 : 0))
//   elems 8..15 = K = kbase + 16..23
__device__ __forceinline__ v16bf ldA(const __bf16* p) {
    union { v16bf v; v4u q[2]; } f;
    f.q[0] = *(const v4u*)(p);
    f.q[1] = *(const v4u*)(p + 16);
    return f.v;
}
// B-fragment (32x16 bf16, KxN): lane L holds col N = L&15,
//   elems 0..7 = K = kk + (L>=16 ? 8 : 0) + 0..7, elems 8..15 = same + 16..23
__device__ __forceinline__ v16bf ldB(const __bf16* p) {
    union { v16bf v; v4u q[2]; } f;
    f.q[0] = *(const v4u*)(p);
    f.q[1] = *(const v4u*)(p + 16);
    return f.v;
}

__device__ __forceinline__ v8f wmma_bf16(v16bf a, v16bf b, v8f c) {
    return __builtin_amdgcn_wmma_f32_16x16x32_bf16(
        /*neg_a=*/false, a, /*neg_b=*/false, b,
        /*c_mod=*/(short)0, c, /*reuse_a=*/false, /*reuse_b=*/false);
}

// ---------------------------------------------------------------------------
// f32 -> bf16 convert
// ---------------------------------------------------------------------------
__global__ void f32_to_bf16_kernel(const float* __restrict__ src,
                                   __bf16* __restrict__ dst, int n) {
    int i = blockIdx.x * 256 + threadIdx.x;
    if (i < n) dst[i] = (__bf16)src[i];
}

// ---------------------------------------------------------------------------
// Generic GEMM: C[M][1024] = A[M][K] * W[N=1024][K]^T (+bias)
// Block tile 128x128, 8 waves, each wave 32(M) x 64(N), K-step 32.
// ---------------------------------------------------------------------------
template <bool OUT_BF16, bool HAS_BIAS>
__global__ __launch_bounds__(256)
void gemm_xwT_kernel(const __bf16* __restrict__ A, const __bf16* __restrict__ W,
                     float* __restrict__ Cf, __bf16* __restrict__ Cb,
                     const float* __restrict__ bias, int K) {
    const int lane = threadIdx.x & 31;
    const int wave = threadIdx.x >> 5;
    const int wm = wave & 3;            // 4 waves along M
    const int wn = wave >> 2;           // 2 waves along N
    const int l15 = lane & 15;
    const int lhi = (lane >> 4) & 1;

    const int m_base = blockIdx.y * 128 + wm * 32;
    const int n_base = blockIdx.x * 128 + wn * 64;

    v8f zero = {};
    v8f acc[2][4];
#pragma unroll
    for (int i = 0; i < 2; i++)
#pragma unroll
        for (int j = 0; j < 4; j++) acc[i][j] = zero;

    const __bf16* Arow0 = A + (size_t)(m_base + l15) * K;
    const __bf16* Arow1 = A + (size_t)(m_base + 16 + l15) * K;
    const __bf16* Wrow[4];
#pragma unroll
    for (int nt = 0; nt < 4; nt++)
        Wrow[nt] = W + (size_t)(n_base + nt * 16 + l15) * K;

    for (int kk = 0; kk < K; kk += 32) {
        __builtin_prefetch(Arow0 + kk + 128, 0, 1);   // -> global_prefetch_b8
        __builtin_prefetch(Wrow[0] + kk + 128, 0, 1);
        v16bf a0 = ldA(Arow0 + kk + lhi * 8);
        v16bf a1 = ldA(Arow1 + kk + lhi * 8);
        const int boff = kk + lhi * 8;
#pragma unroll
        for (int nt = 0; nt < 4; nt++) {
            v16bf bf = ldB(Wrow[nt] + boff);
            acc[0][nt] = wmma_bf16(a0, bf, acc[0][nt]);
            acc[1][nt] = wmma_bf16(a1, bf, acc[1][nt]);
        }
    }
#pragma unroll
    for (int mt = 0; mt < 2; mt++)
#pragma unroll
        for (int nt = 0; nt < 4; nt++)
#pragma unroll
            for (int r = 0; r < 8; r++) {
                int row = m_base + mt * 16 + r + lhi * 8;
                int col = n_base + nt * 16 + l15;
                float v = acc[mt][nt][r];
                if (HAS_BIAS) v += bias[col];
                if (OUT_BF16) Cb[(size_t)row * 1024 + col] = (__bf16)v;
                else          Cf[(size_t)row * 1024 + col] = v;
            }
}

// ---------------------------------------------------------------------------
// V transpose per (b,h): Vt[bh][d][t] = V[(b*1024+t)][h*64+d]
// ---------------------------------------------------------------------------
__global__ void transpose_v_kernel(const __bf16* __restrict__ V,
                                   __bf16* __restrict__ Vt) {
    unsigned i = blockIdx.x * 256 + threadIdx.x;   // 4*1024*1024 elements
    int t  = i & 1023;
    int d  = (i >> 10) & 63;
    int bh = i >> 16;
    int b = bh >> 4, h = bh & 15;
    Vt[i] = V[(size_t)(b * 1024 + t) * 1024 + h * 64 + d];
}

// ---------------------------------------------------------------------------
// Attention: per (b,h, 32-query block). Two-pass softmax via LDS.
//   LDS: Sf = float[32][1024] (128KB), Pb = bf16[32][1024] (64KB)
// ---------------------------------------------------------------------------
__global__ __launch_bounds__(256)
void attn_kernel(const __bf16* __restrict__ Q, const __bf16* __restrict__ Kb,
                 const __bf16* __restrict__ Vt, const float* __restrict__ gate,
                 __bf16* __restrict__ Og) {
    extern __shared__ char smem[];
    float*  Sf = (float*)smem;                              // 32x1024 f32
    __bf16* Pb = (__bf16*)(smem + 32 * 1024 * sizeof(float)); // 32x1024 bf16

    const int lane = threadIdx.x & 31;
    const int wave = threadIdx.x >> 5;
    const int l15 = lane & 15;
    const int lhi = (lane >> 4) & 1;
    const int bh = blockIdx.y;
    const int b = bh >> 4, h = bh & 15;
    const int q0 = blockIdx.x * 32;
    const float gs = 1.0f / (1.0f + __expf(-gate[h]));

    // ---- Phase 1: S = Q K^T / 8, each wave covers 128 key columns ----
    {
        const __bf16* Qrow0 = Q + ((size_t)(b * 1024 + q0 + l15) * 1024 + h * 64);
        const __bf16* Qrow1 = Q + ((size_t)(b * 1024 + q0 + 16 + l15) * 1024 + h * 64);
        v8f zero = {};
#pragma unroll
        for (int nt = 0; nt < 8; nt++) {
            const int n0 = wave * 128 + nt * 16;
            const __bf16* Krow =
                Kb + ((size_t)(b * 1024 + n0 + l15) * 1024 + h * 64);
            v8f acc0 = zero, acc1 = zero;
#pragma unroll
            for (int kk = 0; kk < 64; kk += 32) {
                v16bf a0 = ldA(Qrow0 + kk + lhi * 8);
                v16bf a1 = ldA(Qrow1 + kk + lhi * 8);
                v16bf bf = ldB(Krow + kk + lhi * 8);
                acc0 = wmma_bf16(a0, bf, acc0);
                acc1 = wmma_bf16(a1, bf, acc1);
            }
            const int col = n0 + l15;
#pragma unroll
            for (int r = 0; r < 8; r++) {
                Sf[(r + lhi * 8) * 1024 + col]      = acc0[r] * 0.125f;
                Sf[(16 + r + lhi * 8) * 1024 + col] = acc1[r] * 0.125f;
            }
        }
    }
    __syncthreads();

    // ---- Phase 2: row softmax (sigmoid(gate[h]) folded into P) ----
    for (int rr = 0; rr < 4; rr++) {
        const int row = wave * 4 + rr;
        float mx = -1e30f;
        for (int j = lane; j < 1024; j += 32)
            mx = fmaxf(mx, Sf[row * 1024 + j]);
        for (int off = 16; off; off >>= 1)
            mx = fmaxf(mx, __shfl_xor(mx, off, 32));
        float s = 0.f;
        for (int j = lane; j < 1024; j += 32) {
            float e = __expf(Sf[row * 1024 + j] - mx);
            Sf[row * 1024 + j] = e;
            s += e;
        }
        for (int off = 16; off; off >>= 1) s += __shfl_xor(s, off, 32);
        const float inv = gs / s;
        for (int j = lane; j < 1024; j += 32)
            Pb[row * 1024 + j] = (__bf16)(Sf[row * 1024 + j] * inv);
    }
    __syncthreads();

    // ---- Phase 3: O = P @ V  (one 16x16 tile per wave: mt=wave&1, dt=wave>>1)
    {
        const int mt = wave & 1, dt = wave >> 1;
        const __bf16* Prow = Pb + (size_t)(mt * 16 + l15) * 1024;
        const __bf16* Vrow = Vt + ((size_t)bh * 64 + dt * 16 + l15) * 1024;
        v8f acc = {};
        for (int j0 = 0; j0 < 1024; j0 += 32) {
            v16bf a  = ldA(Prow + j0 + lhi * 8);
            v16bf bf = ldB(Vrow + j0 + lhi * 8);
            acc = wmma_bf16(a, bf, acc);
        }
#pragma unroll
        for (int r = 0; r < 8; r++) {
            int row = b * 1024 + q0 + mt * 16 + r + lhi * 8;
            int col = h * 64 + dt * 16 + l15;
            Og[(size_t)row * 1024 + col] = (__bf16)acc[r];
        }
    }
}

// ---------------------------------------------------------------------------
// h = LayerNorm(X + Y) * g + b, written in-place over Y. One block per row.
// ---------------------------------------------------------------------------
__global__ __launch_bounds__(256)
void add_ln_kernel(const float* __restrict__ X, float* __restrict__ Y,
                   const float* __restrict__ g, const float* __restrict__ bt) {
    __shared__ float red[256];
    const int row = blockIdx.x, tid = threadIdx.x;
    float v[4]; float s = 0.f;
#pragma unroll
    for (int i = 0; i < 4; i++) {
        int e = tid + i * 256;
        v[i] = X[(size_t)row * 1024 + e] + Y[(size_t)row * 1024 + e];
        s += v[i];
    }
    red[tid] = s; __syncthreads();
    for (int off = 128; off; off >>= 1) { if (tid < off) red[tid] += red[tid + off]; __syncthreads(); }
    const float mean = red[0] * (1.f / 1024.f); __syncthreads();
    float vs = 0.f;
#pragma unroll
    for (int i = 0; i < 4; i++) { float d = v[i] - mean; vs += d * d; }
    red[tid] = vs; __syncthreads();
    for (int off = 128; off; off >>= 1) { if (tid < off) red[tid] += red[tid + off]; __syncthreads(); }
    const float rstd = rsqrtf(red[0] * (1.f / 1024.f) + 1e-5f);
#pragma unroll
    for (int i = 0; i < 4; i++) {
        int e = tid + i * 256;
        Y[(size_t)row * 1024 + e] = (v[i] - mean) * rstd * g[e] + bt[e];
    }
}

// out = LayerNorm(H + fvec[broadcast]) * g + b
__global__ __launch_bounds__(256)
void add_vec_ln_kernel(const float* __restrict__ Hm, const float* __restrict__ fvec,
                       const float* __restrict__ g, const float* __restrict__ bt,
                       float* __restrict__ out) {
    __shared__ float red[256];
    const int row = blockIdx.x, tid = threadIdx.x;
    float v[4]; float s = 0.f;
#pragma unroll
    for (int i = 0; i < 4; i++) {
        int e = tid + i * 256;
        v[i] = Hm[(size_t)row * 1024 + e] + fvec[e];
        s += v[i];
    }
    red[tid] = s; __syncthreads();
    for (int off = 128; off; off >>= 1) { if (tid < off) red[tid] += red[tid + off]; __syncthreads(); }
    const float mean = red[0] * (1.f / 1024.f); __syncthreads();
    float vs = 0.f;
#pragma unroll
    for (int i = 0; i < 4; i++) { float d = v[i] - mean; vs += d * d; }
    red[tid] = vs; __syncthreads();
    for (int off = 128; off; off >>= 1) { if (tid < off) red[tid] += red[tid + off]; __syncthreads(); }
    const float rstd = rsqrtf(red[0] * (1.f / 1024.f) + 1e-5f);
#pragma unroll
    for (int i = 0; i < 4; i++) {
        int e = tid + i * 256;
        out[(size_t)row * 1024 + e] = (v[i] - mean) * rstd * g[e] + bt[e];
    }
}

// ---------------------------------------------------------------------------
// Quantum statevector sim: NQ=8 (256 amplitudes), NL=4. One 256-thread block.
// ---------------------------------------------------------------------------
struct c2 { float x, y; };
__device__ __forceinline__ c2 cmul(c2 a, c2 b) { return {a.x * b.x - a.y * b.y, a.x * b.y + a.y * b.x}; }
__device__ __forceinline__ c2 cadd(c2 a, c2 b) { return {a.x + b.x, a.y + b.y}; }

__global__ __launch_bounds__(256)
void quantum_kernel(const float* __restrict__ qp, float* __restrict__ evs) {
    __shared__ c2 amp[256];
    __shared__ float probs[256];
    const int tid = threadIdx.x;
    amp[tid] = {tid == 0 ? 1.f : 0.f, 0.f};
    __syncthreads();
    for (int l = 0; l < 4; l++) {
        for (int q = 0; q < 8; q++) {
            const float t0 = qp[(l * 8 + q) * 3 + 0];
            const float t1 = qp[(l * 8 + q) * 3 + 1];
            const float t2 = qp[(l * 8 + q) * 3 + 2];
            const float cx = cosf(0.5f * t0), sx = sinf(0.5f * t0);
            const float cy = cosf(0.5f * t1), sy = sinf(0.5f * t1);
            const float cz = cosf(0.5f * t2), sz = sinf(0.5f * t2);
            // M = Ry * Rx
            c2 M00 = { cy * cx,  sy * sx};
            c2 M01 = {-sy * cx, -cy * sx};
            c2 M10 = { sy * cx, -cy * sx};
            c2 M11 = { cy * cx, -sy * sx};
            // U = Rz * M,  Rz = diag(cz - i sz, cz + i sz)
            c2 U00 = cmul({cz, -sz}, M00), U01 = cmul({cz, -sz}, M01);
            c2 U10 = cmul({cz,  sz}, M10), U11 = cmul({cz,  sz}, M11);
            const int bp = 7 - q, mask = 1 << bp;
            const int bit = (tid >> bp) & 1;
            c2 a0 = amp[tid & ~mask], a1 = amp[tid | mask];
            __syncthreads();
            amp[tid] = bit ? cadd(cmul(U10, a0), cmul(U11, a1))
                           : cadd(cmul(U00, a0), cmul(U01, a1));
            __syncthreads();
        }
        for (int c = 0; c < 8; c++) {          // CNOT chain, last wraps 7->0
            const int t = (c + 1) & 7;
            const int cb = (tid >> (7 - c)) & 1;
            c2 v = cb ? amp[tid ^ (1 << (7 - t))] : amp[tid];
            __syncthreads();
            amp[tid] = v;
            __syncthreads();
        }
    }
    probs[tid] = amp[tid].x * amp[tid].x + amp[tid].y * amp[tid].y;
    __syncthreads();
    if (tid < 8) {
        float s = 0.f;
        for (int i = 0; i < 256; i++)
            if ((i >> (7 - tid)) & 1) s += probs[i];
        evs[tid] = 1.f - 2.f * s;
    }
}

// hq[4096] = relu(W1 @ evs + bf1)
__global__ void ffn1_kernel(const float* __restrict__ evs, const float* __restrict__ W1,
                            const float* __restrict__ b1, float* __restrict__ hq) {
    int j = blockIdx.x * 256 + threadIdx.x;
    if (j < 4096) {
        float s = b1[j];
#pragma unroll
        for (int q = 0; q < 8; q++) s += W1[j * 8 + q] * evs[q];
        hq[j] = fmaxf(s, 0.f);
    }
}

// fvec[1024] = W2 @ hq + bf2
__global__ void ffn2_kernel(const float* __restrict__ hq, const float* __restrict__ W2,
                            const float* __restrict__ b2, float* __restrict__ fvec) {
    int e = blockIdx.x * 256 + threadIdx.x;
    if (e < 1024) {
        float s = b2[e];
        const float* w = W2 + (size_t)e * 4096;
        for (int j = 0; j < 4096; j++) s += w[j] * hq[j];
        fvec[e] = s;
    }
}

// ---------------------------------------------------------------------------
// Launch
// ---------------------------------------------------------------------------
extern "C" void kernel_launch(void* const* d_in, const int* in_sizes, int n_in,
                              void* d_out, int out_size, void* d_ws, size_t ws_size,
                              hipStream_t stream) {
    const float* x    = (const float*)d_in[0];
    const float* Wq   = (const float*)d_in[1];
    const float* Wk   = (const float*)d_in[2];
    const float* Wv   = (const float*)d_in[3];
    const float* Wo   = (const float*)d_in[4];
    const float* bo   = (const float*)d_in[5];
    const float* gate = (const float*)d_in[6];
    const float* g1   = (const float*)d_in[7];
    const float* b1   = (const float*)d_in[8];
    const float* g2   = (const float*)d_in[9];
    const float* b2   = (const float*)d_in[10];
    // d_in[11]=Wi, d_in[12]=bi : "faithfully computed but unused" in ref; no
    // contribution to the output -> skipped.
    const float* qparams = (const float*)d_in[13];
    const float* W1   = (const float*)d_in[14];
    const float* bf1  = (const float*)d_in[15];
    const float* W2   = (const float*)d_in[16];
    const float* bf2  = (const float*)d_in[17];
    float* out = (float*)d_out;

    char* ws = (char*)d_ws;
    const size_t MiB = 1u << 20;
    __bf16* xb   = (__bf16*)(ws + 0 * MiB);    //  8 MiB  x in bf16
    __bf16* wqb  = (__bf16*)(ws + 8 * MiB);    //  2 MiB
    __bf16* wkb  = (__bf16*)(ws + 10 * MiB);   //  2 MiB
    __bf16* wvb  = (__bf16*)(ws + 12 * MiB);   //  2 MiB
    __bf16* wob  = (__bf16*)(ws + 14 * MiB);   //  2 MiB
    __bf16* Qb   = (__bf16*)(ws + 16 * MiB);   //  8 MiB
    __bf16* Kb   = (__bf16*)(ws + 24 * MiB);   //  8 MiB
    __bf16* Vb   = (__bf16*)(ws + 32 * MiB);   //  8 MiB
    __bf16* Vtb  = (__bf16*)(ws + 40 * MiB);   //  8 MiB  V transposed per (b,h)
    __bf16* outg = (__bf16*)(ws + 48 * MiB);   //  8 MiB  gated attention output
    float*  hbuf = (float*)(ws + 56 * MiB);    // 16 MiB  Wo output -> h (in place)
    float*  evs  = (float*)(ws + 72 * MiB);    //  8 f32
    float*  hq   = (float*)(ws + 72 * MiB + 256);           // 4096 f32
    float*  fvec = (float*)(ws + 72 * MiB + 256 + 16384);   // 1024 f32

    // 1) bf16 converts
    f32_to_bf16_kernel<<<(4096 * 1024) / 256, 256, 0, stream>>>(x, xb, 4096 * 1024);
    f32_to_bf16_kernel<<<(1024 * 1024) / 256, 256, 0, stream>>>(Wq, wqb, 1024 * 1024);
    f32_to_bf16_kernel<<<(1024 * 1024) / 256, 256, 0, stream>>>(Wk, wkb, 1024 * 1024);
    f32_to_bf16_kernel<<<(1024 * 1024) / 256, 256, 0, stream>>>(Wv, wvb, 1024 * 1024);
    f32_to_bf16_kernel<<<(1024 * 1024) / 256, 256, 0, stream>>>(Wo, wob, 1024 * 1024);

    // 2) Q/K/V = x @ W^T  (WMMA bf16)
    dim3 ggrid(1024 / 128, 4096 / 128);
    gemm_xwT_kernel<true, false><<<ggrid, 256, 0, stream>>>(xb, wqb, nullptr, Qb, nullptr, 1024);
    gemm_xwT_kernel<true, false><<<ggrid, 256, 0, stream>>>(xb, wkb, nullptr, Kb, nullptr, 1024);
    gemm_xwT_kernel<true, false><<<ggrid, 256, 0, stream>>>(xb, wvb, nullptr, Vb, nullptr, 1024);

    // 3) transpose V per (b,h)
    transpose_v_kernel<<<(4 * 1024 * 1024) / 256, 256, 0, stream>>>(Vb, Vtb);

    // 4) attention (two-pass softmax in 192 KB LDS)
    const int attn_lds = 32 * 1024 * 4 + 32 * 1024 * 2;  // Sf + Pb
    hipFuncSetAttribute((const void*)attn_kernel,
                        hipFuncAttributeMaxDynamicSharedMemorySize, attn_lds);
    attn_kernel<<<dim3(1024 / 32, 64), 256, attn_lds, stream>>>(Qb, Kb, Vtb, gate, outg);

    // 5) attn_out = outg @ Wo^T + bo  (f32 out)
    gemm_xwT_kernel<false, true><<<ggrid, 256, 0, stream>>>(outg, wob, hbuf, nullptr, bo, 1024);

    // 6) h = LN(x + attn_out)  (in place over hbuf)
    add_ln_kernel<<<4096, 256, 0, stream>>>(x, hbuf, g1, b1);

    // 7) quantum circuit -> evs -> tiny FFN -> broadcast vector fvec
    quantum_kernel<<<1, 256, 0, stream>>>(qparams, evs);
    ffn1_kernel<<<16, 256, 0, stream>>>(evs, W1, bf1, hq);
    ffn2_kernel<<<4, 256, 0, stream>>>(hq, W2, bf2, fvec);

    // 8) out = LN(h + fvec)
    add_vec_ln_kernel<<<4096, 256, 0, stream>>>(hbuf, fvec, g2, b2, out);
}
